// XASNet_GAT_12996571037717
// MI455X (gfx1250) — matrix operand order repaired and checked by reading, not compiled, not run on hardware
//
#include <hip/hip_runtime.h>
#include <hip/hip_bf16.h>
#include <math.h>

// ---------------- problem constants ----------------
#define NN      20000
#define NE      320000
#define D_IN    64
#define HEADS   4
#define CC      128
#define HID     512      // HEADS*CC
#define GG      512
#define LH      128
#define TGT     100
#define NEG_SLOPE 0.2f

typedef __attribute__((ext_vector_type(16))) __bf16 v16bf;
typedef __attribute__((ext_vector_type(8)))  float  v8f;

union FragA { v16bf v; __bf16 e[16]; };
union FragC { v8f   v; float  e[8]; };
union PackBF { __bf16 b[2]; unsigned int u; };

// =====================================================================
// WMMA GEMM:  C[M x Nn] = act( A[M x K] @ B (+bias) ) (+ resid)
//   BT=false: B is K x Nn row-major.   BT=true: B is Nn x K row-major (B^T).
//   ACT: 0 = none, 1 = relu.  HAS_BIAS/HAS_RES: compile-time epilogue flags.
// Tile: 128(M) x 64(N) per block, 8 waves as 4x2, 32x32 per wave (2x2 wmma),
// K stepped by 32.
// LDS holds tiles pre-swizzled into WMMA fragment order:
//   As[sub][lane][elem] : sub = 16-row subtile, 32-byte fragment per lane
//   -> fragment load is a single 32-byte v16bf LDS read, no repacking.
// A-layout (ISA 7.12.2, 16-bit A 16x32): lane<16 K={0-7,16-23}, lane>=16 K={8-15,24-31}
// B-layout (same k order as A, 32x16): lane<16 K={0-7,16-23}, lane>=16 K={8-15,24-31}
// =====================================================================
template<int ACT, bool BT, bool HAS_BIAS, bool HAS_RES>
__global__ __launch_bounds__(256)
void gemm_wmma_bf16(const float* __restrict__ A, const float* __restrict__ B,
                    const float* __restrict__ bias, const float* __restrict__ resid,
                    float* __restrict__ Cmat, int M, int K, int Nn)
{
    __shared__ __bf16 As[8][32][16];   // 8 KB, 32B fragment per (sub,lane)
    __shared__ __bf16 Bs[4][32][16];   // 4 KB

    const int tid  = threadIdx.x;
    const int wave = tid >> 5;
    const int lane = tid & 31;
    const int wm   = wave >> 1;          // 0..3
    const int wn   = wave & 1;           // 0..1
    const int bm   = blockIdx.x * 128;
    const int bn   = blockIdx.y * 64;

    FragC acc[2][2];
    #pragma unroll
    for (int i = 0; i < 2; ++i)
        #pragma unroll
        for (int j = 0; j < 2; ++j)
            #pragma unroll
            for (int v = 0; v < 8; ++v) acc[i][j].e[v] = 0.0f;

    for (int k0 = 0; k0 < K; k0 += 32) {
        // ---- A tile 128x32 as 2048 bf16-pairs; branchless clamped loads ----
        #pragma unroll
        for (int it = 0; it < 8; ++it) {
            int p  = tid + it * 256;         // 0..2047
            int r  = p >> 4;                 // 0..127
            int kp = (p & 15) << 1;          // even k 0..30
            int gr = bm + r;
            int grc = (gr < M) ? gr : (M - 1);
            const float* ap = A + (size_t)grc * K + (k0 + kp);
            float v0 = ap[0];
            float v1 = ap[1];
            if (gr >= M) { v0 = 0.0f; v1 = 0.0f; }
            int sub = r >> 4;
            int ln  = (r & 15) + ((kp & 8) ? 16 : 0);
            int j   = ((kp & 16) >> 1) + (kp & 7);   // even
            PackBF pk; pk.b[0] = (__bf16)v0; pk.b[1] = (__bf16)v1;
            *(unsigned int*)&As[sub][ln][j] = pk.u;
        }
        // ---- B tile 32x64 as 1024 bf16-pairs (paired along K) ----
        #pragma unroll
        for (int it = 0; it < 4; ++it) {
            int p  = tid + it * 256;         // 0..1023
            int n  = p & 63;
            int k  = (p >> 6) << 1;          // even k 0..30
            int gn = bn + n;
            int gnc = (gn < Nn) ? gn : (Nn - 1);
            float v0, v1;
            if (BT) {
                const float* bp = B + (size_t)gnc * K + (k0 + k);
                v0 = bp[0]; v1 = bp[1];
            } else {
                const float* bp = B + (size_t)(k0 + k) * Nn + gnc;
                v0 = bp[0]; v1 = bp[Nn];
            }
            if (gn >= Nn) { v0 = 0.0f; v1 = 0.0f; }
            int sub = n >> 4;
            int ln  = (n & 15) + ((k & 8) ? 16 : 0);
            int j   = ((k & 16) >> 1) + (k & 7);     // even
            PackBF pk; pk.b[0] = (__bf16)v0; pk.b[1] = (__bf16)v1;
            *(unsigned int*)&Bs[sub][ln][j] = pk.u;
        }
        __syncthreads();

        FragA afrag[2], bfrag[2];
        #pragma unroll
        for (int i = 0; i < 2; ++i) afrag[i].v = *(const v16bf*)As[wm * 2 + i][lane];
        #pragma unroll
        for (int i = 0; i < 2; ++i) bfrag[i].v = *(const v16bf*)Bs[wn * 2 + i][lane];

        #pragma unroll
        for (int i = 0; i < 2; ++i)
            #pragma unroll
            for (int j = 0; j < 2; ++j)
                acc[i][j].v = __builtin_amdgcn_wmma_f32_16x16x32_bf16(
                    false, afrag[i].v, false, bfrag[j].v,
                    (short)0, acc[i][j].v, false, false);
        __syncthreads();
    }

    // epilogue: ISA C/D layout (lanes 0-15: M=v, lanes 16-31: M=v+8; N=lane&15)
    #pragma unroll
    for (int i = 0; i < 2; ++i) {
        #pragma unroll
        for (int j = 0; j < 2; ++j) {
            int col   = bn + wn * 32 + j * 16 + (lane & 15);
            int rbase = bm + wm * 32 + i * 16 + ((lane >> 4) << 3);
            if (col < Nn) {
                float bv = HAS_BIAS ? bias[col] : 0.0f;
                float* cp = Cmat + (size_t)rbase * Nn + col;
                const float* rp = HAS_RES ? (resid + (size_t)rbase * Nn + col) : nullptr;
                if (rbase + 7 < M) {
                    // fast path: full 8-row group, branch-free stores
                    #pragma unroll
                    for (int v = 0; v < 8; ++v) {
                        float val = acc[i][j].e[v] + bv;
                        if (ACT == 1) val = fmaxf(val, 0.0f);
                        if (HAS_RES) val += rp[(size_t)v * Nn];
                        cp[(size_t)v * Nn] = val;
                    }
                } else {
                    #pragma unroll
                    for (int v = 0; v < 8; ++v) {
                        if (rbase + v < M) {
                            float val = acc[i][j].e[v] + bv;
                            if (ACT == 1) val = fmaxf(val, 0.0f);
                            if (HAS_RES) val += rp[(size_t)v * Nn];
                            cp[(size_t)v * Nn] = val;
                        }
                    }
                }
            }
        }
    }
}

// ---------------- elementwise / scatter kernels ----------------
__device__ inline float sigmoidf_(float x) { return 1.0f / (1.0f + __expf(-x)); }

__device__ inline void atomicMaxF(float* addr, float val)
{
    int* ai = (int*)addr;
    int old = __float_as_int(*addr);
    while (__int_as_float(old) < val) {
        int assumed = old;
        old = atomicCAS(ai, assumed, __float_as_int(val));
        if (old == assumed) break;
    }
}

__global__ void fill_kernel(float* p, float v, long long n)
{
    long long i = (long long)blockIdx.x * blockDim.x + threadIdx.x;
    if (i < n) p[i] = v;
}

__global__ void gat_scores_kernel(const float* __restrict__ hfeat,
                                  const float* __restrict__ a_src,
                                  const float* __restrict__ a_dst,
                                  float* __restrict__ asrc, float* __restrict__ adst,
                                  int n, int heads, int c)
{
    int i = blockIdx.x * blockDim.x + threadIdx.x;
    if (i >= n * heads) return;
    int node = i / heads, h = i % heads;
    const float* hp = hfeat + (size_t)node * heads * c + (size_t)h * c;
    float s = 0.f, d = 0.f;
    for (int j = 0; j < c; ++j) {
        float v = hp[j];
        s += v * a_src[h * c + j];
        d += v * a_dst[h * c + j];
    }
    asrc[i] = s; adst[i] = d;
}

__global__ void edge_max_kernel(const int* __restrict__ src, const int* __restrict__ dst,
                                const float* __restrict__ asrc, const float* __restrict__ adst,
                                float* __restrict__ maxv, int e, int heads)
{
    int i = blockIdx.x * blockDim.x + threadIdx.x;
    if (i >= e * heads) return;
    int ed = i / heads, h = i % heads;
    int s = src[ed], d = dst[ed];
    float v = asrc[s * heads + h] + adst[d * heads + h];
    v = (v > 0.f) ? v : v * NEG_SLOPE;
    atomicMaxF(&maxv[d * heads + h], v);
}

__global__ void edge_expsum_kernel(const int* __restrict__ src, const int* __restrict__ dst,
                                   const float* __restrict__ asrc, const float* __restrict__ adst,
                                   const float* __restrict__ maxv,
                                   float* __restrict__ ew, float* __restrict__ den,
                                   int e, int heads)
{
    int i = blockIdx.x * blockDim.x + threadIdx.x;
    if (i >= e * heads) return;
    int ed = i / heads, h = i % heads;
    int s = src[ed], d = dst[ed];
    float v = asrc[s * heads + h] + adst[d * heads + h];
    v = (v > 0.f) ? v : v * NEG_SLOPE;
    float w = __expf(v - maxv[d * heads + h]);
    ew[i] = w;
    atomicAdd(&den[d * heads + h], w);
}

__global__ void edge_agg_kernel(const int* __restrict__ src, const int* __restrict__ dst,
                                const float* __restrict__ hfeat,
                                const float* __restrict__ ew, const float* __restrict__ den,
                                float* __restrict__ agg, int e, int heads, int c)
{
    long long total = (long long)e * heads * c;
    long long i = (long long)blockIdx.x * blockDim.x + threadIdx.x;
    if (i >= total) return;
    int hc  = heads * c;
    long long ed = i / hc;
    int rem = (int)(i - ed * hc);
    int h   = rem / c;
    int s = src[ed], d = dst[ed];
    float alpha = ew[ed * heads + h] / (den[d * heads + h] + 1e-16f);
    atomicAdd(&agg[(size_t)d * hc + rem], hfeat[(size_t)s * hc + rem] * alpha);
}

__global__ void bias_act_kernel(const float* __restrict__ in, const float* __restrict__ b,
                                float* __restrict__ out, int n, int cols, int act)
{
    long long i = (long long)blockIdx.x * blockDim.x + threadIdx.x;
    if (i >= (long long)n * cols) return;
    int col = (int)(i % cols);
    float v = in[i] + b[col];
    if (act) v = fmaxf(v, 0.f);
    out[i] = v;
}

__global__ void lstm_gate_kernel(const float* __restrict__ gih, const float* __restrict__ ghh,
                                 float* __restrict__ h, float* __restrict__ c,
                                 float* __restrict__ outbuf /* node stride 256 */, int n)
{
    int i = blockIdx.x * blockDim.x + threadIdx.x;
    if (i >= n * LH) return;
    int node = i / LH, j = i % LH;
    const float* gi = gih + (size_t)node * 4 * LH;
    const float* gh = ghh + (size_t)node * 4 * LH;
    float ig = sigmoidf_(gi[j]          + gh[j]);
    float fg = sigmoidf_(gi[LH + j]     + gh[LH + j]);
    float gg = tanhf    (gi[2 * LH + j] + gh[2 * LH + j]);
    float og = sigmoidf_(gi[3 * LH + j] + gh[3 * LH + j]);
    float cc = fg * c[i] + ig * gg;
    float hh = og * tanhf(cc);
    c[i] = cc; h[i] = hh;
    outbuf[(size_t)node * 256 + j] = hh;
}

__global__ void attn_score_kernel(const float* __restrict__ hfin /* 3 x N x 256 */,
                                  const float* __restrict__ w, const float* __restrict__ b,
                                  float* __restrict__ alpha, int n)
{
    int node = blockIdx.x * blockDim.x + threadIdx.x;
    if (node >= n) return;
    float s[3];
    for (int t = 0; t < 3; ++t) {
        const float* hp = hfin + ((size_t)t * n + node) * 256;
        float acc = 0.f;
        for (int j = 0; j < 256; ++j) acc += hp[j] * w[j];
        s[t] = acc + b[0];
    }
    float m = fmaxf(s[0], fmaxf(s[1], s[2]));
    float e0 = __expf(s[0] - m), e1 = __expf(s[1] - m), e2 = __expf(s[2] - m);
    float inv = 1.0f / (e0 + e1 + e2);
    alpha[node]         = e0 * inv;
    alpha[n + node]     = e1 * inv;
    alpha[2 * n + node] = e2 * inv;
}

__global__ void attn_pool_kernel(const float* __restrict__ x0, const float* __restrict__ x1,
                                 const float* __restrict__ x2, const float* __restrict__ alpha,
                                 float* __restrict__ pooled, int n)
{
    long long i = (long long)blockIdx.x * blockDim.x + threadIdx.x;
    if (i >= (long long)n * HID) return;
    int node = (int)(i / HID);
    pooled[i] = x0[i] * alpha[node] + x1[i] * alpha[n + node] + x2[i] * alpha[2 * n + node];
}

__global__ void seg_sum_kernel(const float* __restrict__ xg, const int* __restrict__ seg,
                               float* __restrict__ gsum, float* __restrict__ cnt, int n)
{
    long long i = (long long)blockIdx.x * blockDim.x + threadIdx.x;
    if (i >= (long long)n * CC) return;
    int node = (int)(i / CC), c = (int)(i % CC);
    int g = seg[node];
    atomicAdd(&gsum[(size_t)g * CC + c], xg[i]);
    if (c == 0) atomicAdd(&cnt[g], 1.0f);
}

__global__ void seg_mean_kernel(const float* __restrict__ gsum, const float* __restrict__ cnt,
                                float* __restrict__ gmean)
{
    int i = blockIdx.x * blockDim.x + threadIdx.x;
    if (i >= GG * CC) return;
    gmean[i] = gsum[i] / fmaxf(cnt[i / CC], 1.0f);
}

// ---------------- host-side orchestration ----------------
static inline void launch_gemm(int act, bool bt, const float* A, const float* B,
                               const float* bias, const float* resid, float* C,
                               int M, int K, int Nn, hipStream_t s)
{
    dim3 g((M + 127) / 128, (Nn + 63) / 64), b(256);
    if (bt) {
        gemm_wmma_bf16<0, true, true, false><<<g, b, 0, s>>>(A, B, bias, resid, C, M, K, Nn);
    } else if (act == 1 && resid) {
        gemm_wmma_bf16<1, false, true, true><<<g, b, 0, s>>>(A, B, bias, resid, C, M, K, Nn);
    } else if (act == 1) {
        gemm_wmma_bf16<1, false, true, false><<<g, b, 0, s>>>(A, B, bias, resid, C, M, K, Nn);
    } else if (bias) {
        gemm_wmma_bf16<0, false, true, false><<<g, b, 0, s>>>(A, B, bias, resid, C, M, K, Nn);
    } else {
        gemm_wmma_bf16<0, false, false, false><<<g, b, 0, s>>>(A, B, bias, resid, C, M, K, Nn);
    }
}

static inline void launch_fill(float* p, float v, long long n, hipStream_t s)
{
    fill_kernel<<<(unsigned)((n + 255) / 256), 256, 0, s>>>(p, v, n);
}

extern "C" void kernel_launch(void* const* d_in, const int* in_sizes, int n_in,
                              void* d_out, int out_size, void* d_ws, size_t ws_size,
                              hipStream_t stream)
{
    (void)in_sizes; (void)n_in; (void)out_size; (void)ws_size;

    const float* x        = (const float*)d_in[0];          // N x 64
    const int*   ei       = (const int*)  d_in[1];          // 2 x E
    const int*   seg      = (const int*)  d_in[2];          // N
    const float* pre_w    = (const float*)d_in[3];          // 64 x 128
    const float* pre_b    = (const float*)d_in[4];
    const float* res_w[4] = {(const float*)d_in[5], (const float*)d_in[6],
                             (const float*)d_in[7], (const float*)d_in[8]};
    const float* res_b[4] = {(const float*)d_in[9], (const float*)d_in[10],
                             (const float*)d_in[11], (const float*)d_in[12]};
    const float* attn_w = (const float*)d_in[53];
    const float* attn_b = (const float*)d_in[54];
    const float* out_w  = (const float*)d_in[55];
    const float* out_b  = (const float*)d_in[56];

    const int* src = ei;
    const int* dst = ei + NE;

    // ---- workspace layout (floats) ----
    float* ws = (float*)d_ws;
    size_t off = 0;
    float* H0A   = ws + off; off += (size_t)NN * CC;
    float* H0B   = ws + off; off += (size_t)NN * CC;
    float* XS[3];
    for (int i = 0; i < 3; ++i) { XS[i] = ws + off; off += (size_t)NN * HID; }
    float* HFEAT = ws + off; off += (size_t)NN * HID;   // also LSTM gih
    float* AGG   = ws + off; off += (size_t)NN * HID;   // also LSTM ghh
    float* ASRC  = ws + off; off += (size_t)NN * HEADS;
    float* ADST  = ws + off; off += (size_t)NN * HEADS;
    float* MAXV  = ws + off; off += (size_t)NN * HEADS;
    float* DEN   = ws + off; off += (size_t)NN * HEADS;
    float* EW    = ws + off; off += (size_t)NE * HEADS;
    float* LOUTA = ws + off; off += (size_t)3 * NN * 256;
    float* LOUTB = ws + off; off += (size_t)3 * NN * 256;   // also POOLED
    float* HST   = ws + off; off += (size_t)NN * LH;
    float* CST   = ws + off; off += (size_t)NN * LH;
    float* ALPHA = ws + off; off += (size_t)3 * NN;
    float* GSUM  = ws + off; off += (size_t)GG * CC;
    float* CNT   = ws + off; off += (size_t)GG;
    float* GMEAN = ws + off; off += (size_t)GG * CC;

    // ================= 1. pre-MLP + 4 residual blocks =================
    launch_gemm(1, false, x, pre_w, pre_b, nullptr, H0A, NN, D_IN, CC, stream);
    float* cur = H0A; float* nxt = H0B;
    for (int i = 0; i < 4; ++i) {
        launch_gemm(1, false, cur, res_w[i], res_b[i], cur, nxt, NN, CC, CC, stream);
        float* t = cur; cur = nxt; nxt = t;
    }

    // ================= 2. three GAT layers (H=4, C=128) =================
    const int eh  = NE * HEADS;
    const int nh  = NN * HEADS;
    for (int g = 0; g < 3; ++g) {
        const float* gw = (const float*)d_in[13 + 4 * g];
        const float* ga = (const float*)d_in[14 + 4 * g];
        const float* gd = (const float*)d_in[15 + 4 * g];
        const float* gb = (const float*)d_in[16 + 4 * g];
        const float* gin = (g == 0) ? cur : XS[g - 1];
        int kin = (g == 0) ? CC : HID;

        launch_gemm(0, false, gin, gw, nullptr, nullptr, HFEAT, NN, kin, HID, stream);
        gat_scores_kernel<<<(nh + 255) / 256, 256, 0, stream>>>(HFEAT, ga, gd, ASRC, ADST, NN, HEADS, CC);
        launch_fill(MAXV, -1e30f, nh, stream);
        launch_fill(DEN, 0.0f, nh, stream);
        launch_fill(AGG, 0.0f, (long long)NN * HID, stream);
        edge_max_kernel<<<(eh + 255) / 256, 256, 0, stream>>>(src, dst, ASRC, ADST, MAXV, NE, HEADS);
        edge_expsum_kernel<<<(eh + 255) / 256, 256, 0, stream>>>(src, dst, ASRC, ADST, MAXV, EW, DEN, NE, HEADS);
        long long tot = (long long)NE * HEADS * CC;
        edge_agg_kernel<<<(unsigned)((tot + 255) / 256), 256, 0, stream>>>(src, dst, HFEAT, EW, DEN, AGG, NE, HEADS, CC);
        bias_act_kernel<<<(unsigned)(((long long)NN * HID + 255) / 256), 256, 0, stream>>>(AGG, gb, XS[g], NN, HID, 1);
    }

    // ================= 3. 3-layer bidirectional LSTM over t=0..2 =================
    float* lin  = nullptr;
    float* lout = LOUTA;
    for (int l = 0; l < 3; ++l) {
        int din = (l == 0) ? HID : 256;
        for (int d = 0; d < 2; ++d) {
            int pb = 29 + (l * 2 + d) * 4;
            const float* w_ih = (const float*)d_in[pb + 0];   // 512 x din
            const float* w_hh = (const float*)d_in[pb + 1];   // 512 x 128
            const float* b_ih = (const float*)d_in[pb + 2];
            const float* b_hh = (const float*)d_in[pb + 3];
            launch_fill(HST, 0.0f, (long long)NN * LH, stream);
            launch_fill(CST, 0.0f, (long long)NN * LH, stream);
            for (int s2 = 0; s2 < 3; ++s2) {
                int t = d ? (2 - s2) : s2;
                const float* xt = (l == 0) ? XS[t] : (lin + (size_t)t * NN * 256);
                launch_gemm(0, true, xt, w_ih, b_ih, nullptr, HFEAT, NN, din, 4 * LH, stream);
                launch_gemm(0, true, HST, w_hh, b_hh, nullptr, AGG, NN, LH, 4 * LH, stream);
                float* ob = lout + (size_t)t * NN * 256 + (size_t)d * LH;
                lstm_gate_kernel<<<(NN * LH + 255) / 256, 256, 0, stream>>>(HFEAT, AGG, HST, CST, ob, NN);
            }
        }
        lin  = lout;
        lout = (lout == LOUTA) ? LOUTB : LOUTA;
    }
    float* HFIN = lin;               // final bi-LSTM output: 3 x N x 256

    // ================= 4. attention pooling over timesteps =================
    attn_score_kernel<<<(NN + 255) / 256, 256, 0, stream>>>(HFIN, attn_w, attn_b, ALPHA, NN);
    float* POOLED = (HFIN == LOUTA) ? LOUTB : LOUTA;
    attn_pool_kernel<<<(unsigned)(((long long)NN * HID + 255) / 256), 256, 0, stream>>>(
        XS[0], XS[1], XS[2], ALPHA, POOLED, NN);

    // ================= 5. output GAT (heads=1, C=128) =================
    {
        const float* gw = (const float*)d_in[25];   // 512 x 128
        const float* ga = (const float*)d_in[26];
        const float* gd = (const float*)d_in[27];
        const float* gb = (const float*)d_in[28];
        launch_gemm(0, false, POOLED, gw, nullptr, nullptr, HFEAT, NN, HID, CC, stream);
        gat_scores_kernel<<<(NN + 255) / 256, 256, 0, stream>>>(HFEAT, ga, gd, ASRC, ADST, NN, 1, CC);
        launch_fill(MAXV, -1e30f, NN, stream);
        launch_fill(DEN, 0.0f, NN, stream);
        launch_fill(AGG, 0.0f, (long long)NN * CC, stream);
        edge_max_kernel<<<(NE + 255) / 256, 256, 0, stream>>>(src, dst, ASRC, ADST, MAXV, NE, 1);
        edge_expsum_kernel<<<(NE + 255) / 256, 256, 0, stream>>>(src, dst, ASRC, ADST, MAXV, EW, DEN, NE, 1);
        long long tot = (long long)NE * CC;
        edge_agg_kernel<<<(unsigned)((tot + 255) / 256), 256, 0, stream>>>(src, dst, HFEAT, EW, DEN, AGG, NE, 1, CC);
        bias_act_kernel<<<(unsigned)(((long long)NN * CC + 255) / 256), 256, 0, stream>>>(AGG, gb, H0A, NN, CC, 0);
    }

    // ================= 6. segment mean pool + output head =================
    launch_fill(GSUM, 0.0f, (long long)GG * CC, stream);
    launch_fill(CNT, 0.0f, GG, stream);
    seg_sum_kernel<<<(unsigned)(((long long)NN * CC + 255) / 256), 256, 0, stream>>>(H0A, seg, GSUM, CNT, NN);
    seg_mean_kernel<<<(GG * CC + 255) / 256, 256, 0, stream>>>(GSUM, CNT, GMEAN);
    launch_gemm(0, false, GMEAN, out_w, out_b, nullptr, (float*)d_out, GG, CC, TGT, stream);
}
